// SniperGRU_35012573397395
// MI455X (gfx1250) — matrix operand order repaired and checked by reading, not compiled, not run on hardware
//
#include <hip/hip_runtime.h>

// GRU: B=512, T=512, F=64, H=128, G=3H=384
// One workgroup owns a 16-row batch tile and runs all T steps internally.
// Wave w owns gate N-tiles {w, 8+w, 16+w}  ==> r, z, n for the SAME 16 hidden
// columns, so gate math happens in-register on the WMMA D-fragments and the
// f32 hidden state lives in a persistent v8f register per wave.
// LDS is only used for the bf16 h hand-off (A-fragments) and W_out partials.
// Biases ride as the C operand of the first WMMA of each chain.

typedef __attribute__((ext_vector_type(16))) __bf16 v16bf;
typedef __attribute__((ext_vector_type(8)))  float  v8f;

static constexpr int Bsz = 512;
static constexpr int Tt  = 512;
static constexpr int Ff  = 64;
static constexpr int Hh  = 128;
static constexpr int BT  = 16;   // batch tile rows per workgroup
static constexpr int HBS = 136;  // bf16 h stride (272B rows: 16B aligned, 4-bank shift)
static constexpr int PS  = 17;   // W_out partial stride (17 floats: odd -> conflict-free)

union FragAB { v16bf v; uint4 q[2]; };

__device__ __forceinline__ float fast_sigmoid(float x) {
    // native v_rcp_f32 instead of IEEE div sequence
    return __builtin_amdgcn_rcpf(1.0f + __expf(-x));
}
__device__ __forceinline__ float fast_tanh(float x) {
    float e = __expf(2.0f * x);
    return 1.0f - 2.0f * __builtin_amdgcn_rcpf(e + 1.0f);
}

// B-fragment from global f32 weights: lane covers column g, K in [kbase, kbase+16).
// VGPR i holds K=kbase+2i (lo16), kbase+2i+1 (hi16).
__device__ __forceinline__ v16bf load_w_frag(const float* __restrict__ W, int ldw,
                                             int g, int kbase) {
    const float4* p = (const float4*)(W + (size_t)g * ldw + kbase);
    float4 a = p[0], b = p[1], c = p[2], d = p[3];
    float f[16] = {a.x,a.y,a.z,a.w, b.x,b.y,b.z,b.w, c.x,c.y,c.z,c.w, d.x,d.y,d.z,d.w};
    v16bf v;
#pragma unroll
    for (int i = 0; i < 16; ++i) v[i] = (__bf16)f[i];
    return v;
}

// A-fragment for x (global f32 -> bf16).
// half=0 lanes: K = k0+0..15; half=1: K = k0+16..31.
__device__ __forceinline__ v16bf load_x_frag(const float* __restrict__ xrow,
                                             int k0, int half) {
    const float4* p = (const float4*)(xrow + k0 + 16*half);
    const float4* q = (const float4*)(xrow + k0 + 8 + 16*half);
    float4 a = p[0], b = p[1], c = q[0], d = q[1];
    float f[16] = {a.x,a.y,a.z,a.w, b.x,b.y,b.z,b.w, c.x,c.y,c.z,c.w, d.x,d.y,d.z,d.w};
    v16bf v;
#pragma unroll
    for (int i = 0; i < 16; ++i) v[i] = (__bf16)f[i];
    return v;
}

__global__ __launch_bounds__(256, 1)
void SniperGRU_gfx1250_kernel(const float* __restrict__ x,
                              const float* __restrict__ Wih,
                              const float* __restrict__ Whh,
                              const float* __restrict__ bih,
                              const float* __restrict__ bhh,
                              const float* __restrict__ Wout,
                              const float* __restrict__ bout,
                              float* __restrict__ out) {
    __shared__ __bf16 sHb[BT * HBS];      // h, bf16, WMMA-A feed (row-major [row][k])
    __shared__ float  sPart[Hh * PS];     // W_out partials, [col][row], stride 17

    const int tid   = threadIdx.x;
    const int lane  = tid & 31;
    const int wave  = tid >> 5;           // 8 waves
    const int l     = lane & 15;
    const int half  = lane >> 4;          // 0: lanes 0-15 (rows 0-7), 1: rows 8-15
    const int bBase = blockIdx.x * BT;

    for (int i = tid; i < BT * HBS; i += 256) sHb[i] = (__bf16)0.0f;

    // Wave w owns hidden columns j in [16w, 16w+16); this lane owns j = 16w + l.
    const int j  = wave * 16 + l;
    const int gr = j;                     // r-gate row in W (tile w)
    const int gz = Hh + j;                // z-gate row (tile 8+w)
    const int gn = 2*Hh + j;              // n-gate row (tile 16+w)

    // Loop-invariant bf16 B-fragments, all register-resident.
    v16bf WHr[4], WHz[4], WHn[4];         // h @ W_hh^T : K=128 -> 4 K-tiles
    v16bf WIr[2], WIz[2], WIn[2];         // x @ W_ih^T : K=64  -> 2 K-tiles
#pragma unroll
    for (int kt = 0; kt < 4; ++kt) {
        WHr[kt] = load_w_frag(Whh, Hh, gr, kt*32 + 16*half);
        WHz[kt] = load_w_frag(Whh, Hh, gz, kt*32 + 16*half);
        WHn[kt] = load_w_frag(Whh, Hh, gn, kt*32 + 16*half);
    }
#pragma unroll
    for (int kt = 0; kt < 2; ++kt) {
        WIr[kt] = load_w_frag(Wih, Ff, gr, kt*32 + 16*half);
        WIz[kt] = load_w_frag(Wih, Ff, gz, kt*32 + 16*half);
        WIn[kt] = load_w_frag(Wih, Ff, gn, kt*32 + 16*half);
    }

    // Persistent splatted bias vectors: C operand of the first WMMA per chain.
    v8f bRv, bZv, bNIv, bNHv;
    {
        const float bR  = bih[gr] + bhh[gr];
        const float bZ  = bih[gz] + bhh[gz];
        const float bNI = bih[gn];
        const float bNH = bhh[gn];
#pragma unroll
        for (int v = 0; v < 8; ++v) {
            bRv[v] = bR; bZv[v] = bZ; bNIv[v] = bNI; bNHv[v] = bNH;
        }
    }
    const float wout_j = Wout[j];
    const float b_out  = bout[0];

    v8f hPrev = {};                        // f32 h for (rows v+8*half, col j)

    __syncthreads();

    // This lane's x row (batch row == WMMA A row == l); step t adds t*F.
    const float* xlane = x + (size_t)(bBase + l) * Tt * Ff;

    for (int t = 0; t < Tt; ++t) {
        const float* xr = xlane + (size_t)t * Ff;
        if (t + 1 < Tt) {                  // gfx1250 global_prefetch_b8
            __builtin_prefetch(xr + Ff, 0, 0);
            __builtin_prefetch(xr + Ff + 32, 0, 0);
        }

        v16bf XA[2], HA[4];
#pragma unroll
        for (int kt = 0; kt < 2; ++kt) XA[kt] = load_x_frag(xr, kt*32, half);
#pragma unroll
        for (int kt = 0; kt < 4; ++kt) {
            FragAB fr;
            const __bf16* base = sHb + l*HBS + kt*32 + 16*half;
            fr.q[0] = *(const uint4*)(base);
            fr.q[1] = *(const uint4*)(base + 8);
            HA[kt] = fr.v;
        }

        // Four accumulator chains; bias enters as C of the first WMMA.
        v8f accR  = __builtin_amdgcn_wmma_f32_16x16x32_bf16(
            false, XA[0], false, WIr[0], (short)0, bRv,  false, false);
        v8f accZ  = __builtin_amdgcn_wmma_f32_16x16x32_bf16(
            false, XA[0], false, WIz[0], (short)0, bZv,  false, false);
        v8f accNI = __builtin_amdgcn_wmma_f32_16x16x32_bf16(
            false, XA[0], false, WIn[0], (short)0, bNIv, false, false);
        accR  = __builtin_amdgcn_wmma_f32_16x16x32_bf16(
            false, XA[1], false, WIr[1], (short)0, accR,  false, false);
        accZ  = __builtin_amdgcn_wmma_f32_16x16x32_bf16(
            false, XA[1], false, WIz[1], (short)0, accZ,  false, false);
        accNI = __builtin_amdgcn_wmma_f32_16x16x32_bf16(
            false, XA[1], false, WIn[1], (short)0, accNI, false, false);

        v8f accNH = __builtin_amdgcn_wmma_f32_16x16x32_bf16(
            false, HA[0], false, WHn[0], (short)0, bNHv, false, false);
        accR  = __builtin_amdgcn_wmma_f32_16x16x32_bf16(
            false, HA[0], false, WHr[0], (short)0, accR,  false, false);
        accZ  = __builtin_amdgcn_wmma_f32_16x16x32_bf16(
            false, HA[0], false, WHz[0], (short)0, accZ,  false, false);
#pragma unroll
        for (int kt = 1; kt < 4; ++kt) {
            accR  = __builtin_amdgcn_wmma_f32_16x16x32_bf16(
                false, HA[kt], false, WHr[kt], (short)0, accR,  false, false);
            accZ  = __builtin_amdgcn_wmma_f32_16x16x32_bf16(
                false, HA[kt], false, WHz[kt], (short)0, accZ,  false, false);
            accNH = __builtin_amdgcn_wmma_f32_16x16x32_bf16(
                false, HA[kt], false, WHn[kt], (short)0, accNH, false, false);
        }

        // Gate math in registers on the D-fragments.
        __bf16* hbp = sHb + j;                       // + row*HBS per element
        float*  ppp = sPart + j*PS + 8*half;         // + v per element
#pragma unroll
        for (int v = 0; v < 8; ++v) {
            const int row = v + 8*half;
            const float r  = fast_sigmoid(accR[v]);
            const float z  = fast_sigmoid(accZ[v]);
            const float nn = fast_tanh(accNI[v] + r * accNH[v]);
            const float h  = fmaf(z, hPrev[v] - nn, nn);   // (1-z)*n + z*h
            hPrev[v] = h;
            hbp[row * HBS] = (__bf16)h;                    // bf16 A-feed
            ppp[v] = h * wout_j;                           // W_out partial
        }
        __syncthreads();

        // Finalize out[b][t]: 16 threads per row sum 8 columns, then butterfly.
        {
            const int row = tid >> 4;
            const int cg  = (tid & 15) * 8;
            float s = 0.0f;
#pragma unroll
            for (int c = 0; c < 8; ++c) s += sPart[(cg + c)*PS + row];
            s += __shfl_xor(s, 1, 32);
            s += __shfl_xor(s, 2, 32);
            s += __shfl_xor(s, 4, 32);
            s += __shfl_xor(s, 8, 32);
            if ((tid & 15) == 0)
                out[(size_t)(bBase + row) * Tt + t] = fast_tanh(s + b_out);
        }
        __syncthreads();
    }
}

extern "C" void kernel_launch(void* const* d_in, const int* in_sizes, int n_in,
                              void* d_out, int out_size, void* d_ws, size_t ws_size,
                              hipStream_t stream) {
    (void)in_sizes; (void)n_in; (void)d_ws; (void)ws_size; (void)out_size;
    const float* x    = (const float*)d_in[0];
    const float* Wih  = (const float*)d_in[1];
    const float* Whh  = (const float*)d_in[2];
    const float* bih  = (const float*)d_in[3];
    const float* bhh  = (const float*)d_in[4];
    const float* Wout = (const float*)d_in[5];
    const float* bout = (const float*)d_in[6];
    float* out = (float*)d_out;

    dim3 grid(Bsz / BT);   // 32 workgroups
    dim3 block(256);       // 8 wave32 waves
    SniperGRU_gfx1250_kernel<<<grid, block, 0, stream>>>(
        x, Wih, Whh, bih, bhh, Wout, bout, out);
}
